// SwinBlock_37838661878648
// MI455X (gfx1250) — matrix operand order repaired and checked by reading, not compiled, not run on hardware
//
#include <hip/hip_runtime.h>
#include <hip/hip_bf16.h>

typedef __attribute__((ext_vector_type(16))) _Float16 v16h;
typedef __attribute__((ext_vector_type(8)))  float    v8f;
typedef __attribute__((ext_vector_type(4)))  unsigned int v4u;
typedef __attribute__((ext_vector_type(8)))  int      v8i;
typedef __attribute__((ext_vector_type(4)))  int      v4i;

#define DIMC  192
#define NHEAD 6
#define HD    32
#define NTOK  64          // tokens per 8x8 window
#define NB    8
#define LTOT  4096        // 64*64
#define MTOT  32768       // NB*LTOT
#define NWIN  512
#define NEGV  (-100.0f)
#define KCHUNK 192

// ---------------- WMMA fragment loaders (per CDNA5 ISA VGPR layouts) ----------------
// A (16x32 f16): lanes 0-15 hold row M=lane, K = {0..7, 16..23}; lanes 16-31 hold
// row M=lane-16, K = {8..15, 24..31}.  -> two 16B loads at +s and +s+16 (s=0 or 8).
// B (32x16 f16): lanes 0-15 hold col N=lane, K = {0..7, 16..23}; lanes 16-31 col N=lane-16,
// K = {8..15, 24..31}. With B stored as Bcol[N][K]: two 16B loads at +s and +s+16 (s=0 or 8).
union F16x16 { v16h v; float4 q[2]; };

__device__ __forceinline__ v16h ldA16(const _Float16* p) {
  F16x16 u;
  u.q[0] = *reinterpret_cast<const float4*>(p);
  u.q[1] = *reinterpret_cast<const float4*>(p + 16);
  return u.v;
}
__device__ __forceinline__ v16h ldB16(const _Float16* p) {
  F16x16 u;
  u.q[0] = *reinterpret_cast<const float4*>(p);
  u.q[1] = *reinterpret_cast<const float4*>(p + 16);
  return u.v;
}

// ---------------- TDM: async 2D f16 tile load Global -> LDS ----------------
// Builds a Tensor DMA Descriptor (D#) per CDNA5 ISA:
//  group0: [1:0]count=1, [63:32]lds_addr, [120:64]global_addr, [127:126]type=2
//  group1: [17:16]data_size=1 (2B), [79:48]tensor_dim0, [111:80]tensor_dim1,
//          [127:112]tile_dim0, [143:128]tile_dim1, [207:160]tensor_dim0_stride
// Tile is packed densely into LDS (row length = cols elements). 2D: groups 2/3 zero.
__device__ __forceinline__ void tdm_load_2d_f16(unsigned lds_byte_off,
                                                const _Float16* gsrc,
                                                unsigned rows, unsigned cols,
                                                unsigned stride_elems) {
  unsigned long long ga = (unsigned long long)(uintptr_t)gsrc;
  v4u g0;
  g0.x = 1u;                                            // count=1 (valid user D#)
  g0.y = lds_byte_off;                                  // lds_addr
  g0.z = (unsigned)(ga & 0xFFFFFFFFu);                  // global_addr[31:0]
  g0.w = (unsigned)((ga >> 32) & 0x01FFFFFFu) | (2u << 30); // addr[56:32] | type=2
  v8i g1;
  g1[0] = (int)(1u << 16);                              // data_size=1 (2 bytes)
  g1[1] = (int)((cols & 0xFFFFu) << 16);                // tensor_dim0[15:0]
  g1[2] = (int)(((cols >> 16) & 0xFFFFu) | ((rows & 0xFFFFu) << 16)); // td0[31:16]|td1[15:0]
  g1[3] = (int)(((rows >> 16) & 0xFFFFu) | ((cols & 0xFFFFu) << 16)); // td1[31:16]|tile_dim0
  g1[4] = (int)(rows & 0xFFFFu);                        // tile_dim1 (tile_dim2=0)
  g1[5] = (int)stride_elems;                            // tensor_dim0_stride[31:0]
  g1[6] = 0;                                            // stride[47:32]=0, d1stride lo=0
  g1[7] = 0;
  v4i z4 = {0, 0, 0, 0};
#if defined(__clang_major__) && (__clang_major__ >= 23)
  v8i z8 = {0, 0, 0, 0, 0, 0, 0, 0};
  __builtin_amdgcn_tensor_load_to_lds(g0, g1, z4, z4, z8, 0);
#else
  __builtin_amdgcn_tensor_load_to_lds(g0, g1, z4, z4, 0);
#endif
}

// ---------------- prep kernels ----------------
__global__ void cvt_f16_kernel(const float* __restrict__ s, _Float16* __restrict__ d, int n) {
  int i = blockIdx.x * blockDim.x + threadIdx.x;
  if (i < n) d[i] = (_Float16)s[i];
}

// rpb_full[h][m][n] = rpb_table[((r1-r2+7)*15 + (c1-c2+7))*6 + h]
__global__ void rpb_kernel(const float* __restrict__ tab, float* __restrict__ out) {
  int i = blockIdx.x * blockDim.x + threadIdx.x;   // 6*4096 exact
  int h = i >> 12;
  int mn = i & 4095;
  int m = mn >> 6, n = mn & 63;
  int r1 = m >> 3, c1 = m & 7, r2 = n >> 3, c2 = n & 7;
  int idx = (r1 - r2 + 7) * 15 + (c1 - c2 + 7);
  out[i] = tab[idx * NHEAD + h];
}

// ---------------- LayerNorm 1: x (B,C,H,W) -> shortcut fp32 (B,L,C) + xin f16 (window order) ----------------
__global__ void __launch_bounds__(256) ln1_kernel(const float* __restrict__ x,
                                                  const float* __restrict__ g,
                                                  const float* __restrict__ bta,
                                                  float* __restrict__ shortcut,
                                                  _Float16* __restrict__ xin) {
  int tokL = blockIdx.x * blockDim.x + threadIdx.x;   // b*4096 + l, exact MTOT
  int b = tokL >> 12, l = tokL & 4095;
  const float* xp = x + (size_t)b * DIMC * LTOT + l;
  float s = 0.f, ss = 0.f;
  for (int c = 0; c < DIMC; ++c) {
    float v = xp[(size_t)c * LTOT];
    s += v; ss += v * v;
  }
  float mu = s * (1.0f / DIMC);
  float var = ss * (1.0f / DIMC) - mu * mu;
  float rstd = rsqrtf(fmaxf(var, 0.f) + 1e-5f);
  // window mapping: l = (hb*8+r)*64 + (wb*8+cw)
  int hh = l >> 6, wwp = l & 63;
  int hb = hh >> 3, r = hh & 7, wb = wwp >> 3, cw = wwp & 7;
  int win = (b << 6) + (hb << 3) + wb;
  int tok = (r << 3) + cw;
  float* sp = shortcut + (size_t)tokL * DIMC;
  _Float16* xo = xin + ((size_t)(win << 6) + tok) * DIMC;
  for (int c = 0; c < DIMC; ++c) {
    float v = (xp[(size_t)c * LTOT] - mu) * rstd * g[c] + bta[c];
    sp[c] = v;
    xo[c] = (_Float16)v;
  }
}

// ---------------- LayerNorm 2: x2 (M,C) fp32 -> xn f16 ----------------
__global__ void __launch_bounds__(256) ln2_kernel(const float* __restrict__ x2,
                                                  const float* __restrict__ g,
                                                  const float* __restrict__ bta,
                                                  _Float16* __restrict__ xn) {
  int m = blockIdx.x * blockDim.x + threadIdx.x;
  const float* p = x2 + (size_t)m * DIMC;
  float s = 0.f, ss = 0.f;
  for (int c = 0; c < DIMC; ++c) { float v = p[c]; s += v; ss += v * v; }
  float mu = s * (1.0f / DIMC);
  float var = ss * (1.0f / DIMC) - mu * mu;
  float rstd = rsqrtf(fmaxf(var, 0.f) + 1e-5f);
  _Float16* o = xn + (size_t)m * DIMC;
  for (int c = 0; c < DIMC; ++c)
    o[c] = (_Float16)((p[c] - mu) * rstd * g[c] + bta[c]);
}

// ---------------- TDM-staged WMMA GEMM: C[M,N] = A[M,K] * Bcol[N,K]^T + bias ----------------
// Block = 128 threads (4 waves) computes a 16(M) x 64(N) tile.
// Per K-chunk of 192: wave 0 TDM-loads A panel (16xKc) and B panel (64xKc) into LDS,
// waits TENSORcnt==0, barrier; each wave runs 6 WMMAs for its own 16-wide N tile,
// sharing the A panel across all 4 waves.
// EPI 0: store fp32 [m][n]                (qkv)
// EPI 1: window-reverse + shortcut resid  (proj -> x2)
// EPI 2: exact GELU -> f16 [m][n]         (fc1)
// EPI 3: + x2 resid, transposed NCHW out  (fc2)
template<int EPI>
__global__ void __launch_bounds__(128) wmma_gemm(const _Float16* __restrict__ A,
                                                 const _Float16* __restrict__ Bc,
                                                 const float* __restrict__ bias,
                                                 int N, int K,
                                                 float* __restrict__ outF,
                                                 _Float16* __restrict__ outH,
                                                 const float* __restrict__ extra) {
  __shared__ __align__(16) _Float16 As[16 * KCHUNK];   // 6 KB
  __shared__ __align__(16) _Float16 Bs[64 * KCHUNK];   // 24 KB
  const int tid  = threadIdx.x;
  const int wid  = tid >> 5;
  const int lane = tid & 31;
  const int tmBase = blockIdx.y << 4;   // 16 M-rows per block
  const int tnBase = blockIdx.x << 6;   // 64 N-cols per block

  v8f acc = {0.f, 0.f, 0.f, 0.f, 0.f, 0.f, 0.f, 0.f};
  const _Float16* Ap = As + (lane & 15) * KCHUNK + ((lane < 16) ? 0 : 8);
  const _Float16* Bp = Bs + ((wid << 4) + (lane & 15)) * KCHUNK + ((lane < 16) ? 0 : 8);

  for (int kc = 0; kc < K; kc += KCHUNK) {
    if (kc) __syncthreads();     // previous chunk fully consumed before overwrite
    if (tid < 32) {              // wave 0 issues the DMAs (TDM ignores EXEC)
      tdm_load_2d_f16((unsigned)(uintptr_t)As, A + (size_t)tmBase * K + kc, 16, KCHUNK, (unsigned)K);
      tdm_load_2d_f16((unsigned)(uintptr_t)Bs, Bc + (size_t)tnBase * K + kc, 64, KCHUNK, (unsigned)K);
      __builtin_amdgcn_s_wait_tensorcnt((short)0);
    }
    __syncthreads();
    #pragma unroll
    for (int kk = 0; kk < KCHUNK; kk += 32) {
      v16h a = ldA16(Ap + kk);
      v16h b = ldB16(Bp + kk);
      acc = __builtin_amdgcn_wmma_f32_16x16x32_f16(false, a, false, b, (short)0, acc, false, false);
    }
  }

  int n  = tnBase + (wid << 4) + (lane & 15);
  int mb = tmBase + ((lane >> 4) << 3);
  float bv = bias[n];
  #pragma unroll
  for (int r = 0; r < 8; ++r) {
    int m = mb + r;
    float val = acc[r] + bv;
    if (EPI == 0) {
      outF[(size_t)m * N + n] = val;
    } else if (EPI == 1) {
      int win = m >> 6, tok = m & 63;
      int b_ = win >> 6, hb = (win >> 3) & 7, wb = win & 7;
      int rr = tok >> 3, cw = tok & 7;
      int l = (((hb << 3) + rr) << 6) + (wb << 3) + cw;
      size_t idx = ((size_t)(b_ * LTOT + l)) * DIMC + n;
      outF[idx] = extra[idx] + val;
    } else if (EPI == 2) {
      float ge = 0.5f * val * (1.0f + erff(val * 0.7071067811865475f));
      outH[(size_t)m * N + n] = (_Float16)ge;
    } else {
      int b_ = m >> 12, l = m & 4095;
      outF[((size_t)(b_ * DIMC + n)) * LTOT + l] = extra[(size_t)m * DIMC + n] + val;
    }
  }
}

// ---------------- attention: one block per (window, head) ----------------
__global__ void __launch_bounds__(128) attn_kernel(const float* __restrict__ qkv_raw,
                                                   const float* __restrict__ temp,
                                                   const float* __restrict__ rpb_full,
                                                   _Float16* __restrict__ Oc) {
  int win  = blockIdx.x / NHEAD;
  int hset = blockIdx.x % NHEAD;
  __shared__ __align__(16) _Float16 qh[NTOK][HD];   // q l2-normed * temperature
  __shared__ __align__(16) _Float16 kh[NTOK][HD];   // k l2-normed
  __shared__ __align__(16) _Float16 vT[HD][NTOK];   // V transposed (Bcol layout)
  __shared__ __align__(16) float    S[NTOK][NTOK];
  __shared__ __align__(16) _Float16 P[NTOK][NTOK];

  int tid = threadIdx.x;
  if (tid < NTOK) {
    const float* base = qkv_raw + ((size_t)(win * NTOK + tid)) * (3 * DIMC) + hset * HD;
    float sq = 0.f, sk = 0.f;
    for (int d = 0; d < HD; ++d) {
      float qv = base[d], kv = base[DIMC + d];
      sq += qv * qv; sk += kv * kv;
    }
    float qs = temp[hset] / fmaxf(sqrtf(sq), 1e-12f);
    float ks = 1.0f / fmaxf(sqrtf(sk), 1e-12f);
    for (int d = 0; d < HD; ++d) {
      qh[tid][d] = (_Float16)(base[d] * qs);
      kh[tid][d] = (_Float16)(base[DIMC + d] * ks);
      vT[d][tid] = (_Float16)(base[2 * DIMC + d]);
    }
  }
  __syncthreads();

  int wid = tid >> 5, lane = tid & 31;
  // S = qh (64x32) @ kh^T : 4x4 tiles, wave w owns row-stripe tm=w, K=32 -> 1 WMMA/tile
  {
    int tm = wid;
    #pragma unroll
    for (int tn = 0; tn < 4; ++tn) {
      v16h a = ldA16(&qh[tm * 16 + (lane & 15)][(lane < 16) ? 0 : 8]);
      v16h b = ldB16(&kh[tn * 16 + (lane & 15)][(lane < 16) ? 0 : 8]);
      v8f c = {0.f, 0.f, 0.f, 0.f, 0.f, 0.f, 0.f, 0.f};
      c = __builtin_amdgcn_wmma_f32_16x16x32_f16(false, a, false, b, (short)0, c, false, false);
      int n  = tn * 16 + (lane & 15);
      int mb = tm * 16 + ((lane >> 4) << 3);
      #pragma unroll
      for (int r = 0; r < 8; ++r) S[mb + r][n] = c[r];
    }
  }
  __syncthreads();

  // per-row top-16 threshold (exact 16th order statistic), + rpb, softmax -> P f16
  if (tid < NTOK) {
    int m = tid;
    float arr[NTOK], work[NTOK], ex[NTOK];
    for (int n = 0; n < NTOK; ++n) { arr[n] = S[m][n]; work[n] = arr[n]; }
    float thresh = -1e30f;
    for (int it = 0; it < 16; ++it) {
      int bi = 0; float bvv = -1e30f;
      for (int n = 0; n < NTOK; ++n)
        if (work[n] > bvv) { bvv = work[n]; bi = n; }
      work[bi] = -1e30f;
      thresh = bvv;
    }
    const float* rpb = rpb_full + ((size_t)hset * NTOK + m) * NTOK;
    float mx = -1e30f;
    for (int n = 0; n < NTOK; ++n) {
      float v = ((arr[n] >= thresh) ? arr[n] : NEGV) + rpb[n];
      ex[n] = v;
      mx = fmaxf(mx, v);
    }
    float sum = 0.f;
    for (int n = 0; n < NTOK; ++n) { float e = __expf(ex[n] - mx); ex[n] = e; sum += e; }
    float inv = 1.0f / sum;
    for (int n = 0; n < NTOK; ++n) P[m][n] = (_Float16)(ex[n] * inv);
  }
  __syncthreads();

  // O = P (64x64) @ V (64x32): 4x2 tiles, K-loop of 2; write f16 O_concat[tok][h*32+d]
  {
    int tm = wid;
    #pragma unroll
    for (int tn = 0; tn < 2; ++tn) {
      v8f c = {0.f, 0.f, 0.f, 0.f, 0.f, 0.f, 0.f, 0.f};
      #pragma unroll
      for (int k0 = 0; k0 < NTOK; k0 += 32) {
        v16h a = ldA16(&P[tm * 16 + (lane & 15)][k0 + ((lane < 16) ? 0 : 8)]);
        v16h b = ldB16(&vT[tn * 16 + (lane & 15)][k0 + ((lane < 16) ? 0 : 8)]);
        c = __builtin_amdgcn_wmma_f32_16x16x32_f16(false, a, false, b, (short)0, c, false, false);
      }
      int n  = tn * 16 + (lane & 15);         // d index 0..31
      int mb = tm * 16 + ((lane >> 4) << 3);
      #pragma unroll
      for (int r = 0; r < 8; ++r)
        Oc[((size_t)(win * NTOK) + mb + r) * DIMC + hset * HD + n] = (_Float16)c[r];
    }
  }
}

// ---------------- host launch ----------------
extern "C" void kernel_launch(void* const* d_in, const int* in_sizes, int n_in,
                              void* d_out, int out_size, void* d_ws, size_t ws_size,
                              hipStream_t stream) {
  const float* x      = (const float*)d_in[0];
  const float* n1g    = (const float*)d_in[1];
  const float* n1b    = (const float*)d_in[2];
  const float* qkv_w  = (const float*)d_in[3];
  const float* qkv_b  = (const float*)d_in[4];
  const float* proj_w = (const float*)d_in[5];
  const float* proj_b = (const float*)d_in[6];
  const float* rpb_t  = (const float*)d_in[7];
  const float* temp   = (const float*)d_in[8];
  const float* n2g    = (const float*)d_in[9];
  const float* n2b    = (const float*)d_in[10];
  const float* fc1_w  = (const float*)d_in[11];
  const float* fc1_b  = (const float*)d_in[12];
  const float* fc2_w  = (const float*)d_in[13];
  const float* fc2_b  = (const float*)d_in[14];

  char* ws = (char*)d_ws;
  // region map (bytes, time-aliased):
  const size_t OFF_SHORTCUT = 0;            // fp32 32768*192        (live: ln1 .. proj)
  const size_t OFF_B        = 25165824;     // 75.5MB multi-use region
  //   qkv_raw fp32 32768*576 (gemm0 .. attn)
  //   xn f16 at B+0 (ln2 .. fc1), h1 f16 at B+16MB (fc1 .. fc2)
  const size_t OFF_XIN      = 100663296;    // f16 32768*192 (ln1 .. gemm0); reused as O_concat (attn .. proj)
  const size_t OFF_X2       = 113246208;    // fp32 32768*192 (proj .. fc2)
  const size_t OFF_WQ       = 138412032;    // f16 weights
  const size_t OFF_WP       = OFF_WQ + 576 * 192 * 2;
  const size_t OFF_W1       = OFF_WP + 192 * 192 * 2;
  const size_t OFF_W2       = OFF_W1 + 768 * 192 * 2;
  const size_t OFF_RPB      = OFF_W2 + 192 * 768 * 2;

  float*    shortcut = (float*)(ws + OFF_SHORTCUT);
  float*    qkv_raw  = (float*)(ws + OFF_B);
  _Float16* xn       = (_Float16*)(ws + OFF_B);
  _Float16* h1       = (_Float16*)(ws + OFF_B + 16777216);
  _Float16* xin      = (_Float16*)(ws + OFF_XIN);
  _Float16* Oc       = (_Float16*)(ws + OFF_XIN);
  float*    x2       = (float*)(ws + OFF_X2);
  _Float16* wq       = (_Float16*)(ws + OFF_WQ);
  _Float16* wp       = (_Float16*)(ws + OFF_WP);
  _Float16* w1       = (_Float16*)(ws + OFF_W1);
  _Float16* w2       = (_Float16*)(ws + OFF_W2);
  float*    rpbf     = (float*)(ws + OFF_RPB);

  // weight conversion + rpb gather
  cvt_f16_kernel<<<(110592 + 255) / 256, 256, 0, stream>>>(qkv_w, wq, 110592);
  cvt_f16_kernel<<<(36864  + 255) / 256, 256, 0, stream>>>(proj_w, wp, 36864);
  cvt_f16_kernel<<<(147456 + 255) / 256, 256, 0, stream>>>(fc1_w, w1, 147456);
  cvt_f16_kernel<<<(147456 + 255) / 256, 256, 0, stream>>>(fc2_w, w2, 147456);
  rpb_kernel<<<96, 256, 0, stream>>>(rpb_t, rpbf);

  // LN1 -> shortcut + window-ordered f16 activations
  ln1_kernel<<<MTOT / 256, 256, 0, stream>>>(x, n1g, n1b, shortcut, xin);

  // QKV GEMM: 32768 x 576 x 192 (TDM-staged)   grid: N/64 x M/16
  wmma_gemm<0><<<dim3(9, 2048), 128, 0, stream>>>(xin, wq, qkv_b, 576, 192, qkv_raw, nullptr, nullptr);

  // attention per (window, head)
  attn_kernel<<<NWIN * NHEAD, 128, 0, stream>>>(qkv_raw, temp, rpbf, Oc);

  // proj GEMM + window reverse + shortcut residual -> x2
  wmma_gemm<1><<<dim3(3, 2048), 128, 0, stream>>>(Oc, wp, proj_b, 192, 192, x2, nullptr, shortcut);

  // LN2 -> xn f16
  ln2_kernel<<<MTOT / 256, 256, 0, stream>>>(x2, n2g, n2b, xn);

  // fc1 GEMM + GELU -> h1 f16
  wmma_gemm<2><<<dim3(12, 2048), 128, 0, stream>>>(xn, w1, fc1_b, 768, 192, nullptr, h1, nullptr);

  // fc2 GEMM + residual + NCHW transpose -> d_out  (K=768 -> 4 TDM-staged chunks)
  wmma_gemm<3><<<dim3(3, 2048), 128, 0, stream>>>(h1, w2, fc2_b, 192, 768, (float*)d_out, nullptr, x2);
}
